// DeepGCN_53446573031764
// MI455X (gfx1250) — matrix operand order repaired and checked by reading, not compiled, not run on hardware
//
#include <hip/hip_runtime.h>
#include <hip/hip_bf16.h>

// ---------------------------------------------------------------------------
// CDNA5 (gfx1250) DeepGCN forward. All matmuls via v_wmma_f32_16x16x32_bf16.
// Distance strips live entirely in LDS (256KB/WGP) -> no HBM dist matrices.
// Top-k selection uses register-resident per-lane sorted buffers (QD deep),
// so each selection step is a wave-argmin over 32 buffer heads instead of a
// full 4096-element rescan.
// ---------------------------------------------------------------------------

typedef __bf16 bf16_t;
typedef __attribute__((ext_vector_type(16))) __bf16 v16bf;
typedef __attribute__((ext_vector_type(8)))  __bf16 v8bf;
typedef __attribute__((ext_vector_type(8)))  float  v8f;

#define NPTS_   4096
#define NTOT_   16384   // B*NPTS
#define C_      64
#define K_      16
#define NB_     7
#define FEATD_  512     // C*(NB+1)
#define BIGF    3.4e38f

// sorted-ascending register ladder insert (fully unrolled, constant indices)
#define TOPK_INSERT(vv, jj)                                                 \
    {                                                                       \
        float _cv = (vv); int _cj = (jj);                                   \
        _Pragma("unroll")                                                   \
        for (int _q = 0; _q < QD; ++_q) {                                   \
            bool _sw = (_cv < lv[_q]) || (_cv == lv[_q] && _cj < li[_q]);   \
            float _tv = _sw ? lv[_q] : _cv; int _tj = _sw ? li[_q] : _cj;   \
            lv[_q] = _sw ? _cv : lv[_q];    li[_q] = _sw ? _cj : li[_q];    \
            _cv = _tv; _cj = _tj;                                           \
        }                                                                   \
    }

// Select the L smallest of row[0..npts) (one wave; lane owns j==lane mod 32),
// writing every dil-th winner's index to out[0..L/dil).
template <int QD>
__device__ __forceinline__ void topk_select(volatile float* row, int npts, int lane,
                                            int L, int dil, int* out) {
    float lv[QD]; int li[QD];
#pragma unroll
    for (int q = 0; q < QD; ++q) { lv[q] = BIGF; li[q] = 0x7fffffff; }
    for (int j = lane; j < npts; j += 32) {
        float v = row[j];
        if (v < lv[QD - 1]) TOPK_INSERT(v, j);
    }
    int cnt = 0;
#pragma unroll
    for (int q = 0; q < QD; ++q) cnt += (lv[q] < 1e37f) ? 1 : 0;

    for (int it = 0; it < L; ++it) {
        float bv = lv[0]; int bi = li[0];
#pragma unroll
        for (int off = 16; off; off >>= 1) {
            float ov = __shfl_xor(bv, off);
            int   oi = __shfl_xor(bi, off);
            if (ov < bv || (ov == bv && oi < bi)) { bv = ov; bi = oi; }
        }
        if (lane == 0 && (it % dil) == 0) out[it / dil] = bi;
        if (lane == (bi & 31)) {                 // owner pops its head
            row[bi] = BIGF;                      // mark removed for rescans
#pragma unroll
            for (int q = 0; q < QD - 1; ++q) { lv[q] = lv[q + 1]; li[q] = li[q + 1]; }
            lv[QD - 1] = BIGF; li[QD - 1] = 0x7fffffff;
            if (--cnt == 0) {                    // refill from remaining set
                for (int j = lane; j < npts; j += 32) {
                    float v = row[j];
                    if (v < lv[QD - 1]) TOPK_INSERT(v, j);
                }
#pragma unroll
                for (int q = 0; q < QD; ++q) cnt += (lv[q] < 1e37f) ? 1 : 0;
            }
        }
    }
}

// ---------------- weight prep: dst[o*Kpad+k] = (k<Ksrc && o<Nsrc)? src[k*Nsrc+o] : 0
__global__ void k_prep_wt(const float* __restrict__ src, bf16_t* __restrict__ dst,
                          int Ksrc, int Kpad, int Nsrc, int Npad) {
    int t = blockIdx.x * blockDim.x + threadIdx.x;
    if (t >= Npad * Kpad) return;
    int o = t / Kpad, k = t % Kpad;
    float v = (k < Ksrc && o < Nsrc) ? src[k * Nsrc + o] : 0.0f;
    dst[o * Kpad + k] = (bf16_t)v;
}

// ---------------- squared norms of bf16 feature rows
__global__ void k_sqnorm(const bf16_t* __restrict__ h, float* __restrict__ sqn) {
    int p = blockIdx.x * blockDim.x + threadIdx.x;
    if (p >= NTOT_) return;
    float s = 0.f;
    for (int c = 0; c < C_; ++c) { float v = (float)h[p * C_ + c]; s += v * v; }
    sqn[p] = s;
}

// ---------------- f32 -> bf16 convert
__global__ void k_f2bf(const float* __restrict__ src, bf16_t* __restrict__ dst, int n) {
    int t = blockIdx.x * blockDim.x + threadIdx.x;
    if (t < n) dst[t] = (bf16_t)src[t];
}

// ---------------- head KNN: 3-D positions, one wave per query, top-16
__global__ __launch_bounds__(32) void k_knn_head(const float* __restrict__ pos,
                                                 int* __restrict__ idxout) {
    __shared__ float dist[NPTS_];
    int q = blockIdx.x;           // global point id
    int b = q >> 12;
    int base = b << 12;
    int lane = threadIdx.x;
    float qx = pos[q * 3 + 0], qy = pos[q * 3 + 1], qz = pos[q * 3 + 2];
    for (int j = lane; j < NPTS_; j += 32) {
        float dx = pos[(base + j) * 3 + 0] - qx;
        float dy = pos[(base + j) * 3 + 1] - qy;
        float dz = pos[(base + j) * 3 + 2] - qz;
        dist[j] = dx * dx + dy * dy + dz * dz;
    }
    __syncthreads();
    topk_select<4>(dist, NPTS_, lane, K_, 1, idxout + q * K_);
}

// ---------------- feature KNN: Gram via WMMA, 16 queries x 4096 strip in LDS
__global__ __launch_bounds__(512) void k_knn_feat(const bf16_t* __restrict__ h,
                                                  const float* __restrict__ sqn,
                                                  int* __restrict__ idxout, int dil) {
    extern __shared__ float strip[];              // 16 * 4096 floats = 256 KB
    int b    = blockIdx.x >> 8;                   // 256 query-tiles per batch
    int q0   = (blockIdx.x & 255) * 16;
    int lane = threadIdx.x & 31, wv = threadIdx.x >> 5;
    int m = lane & 15, half = lane >> 4, n = lane & 15;
    int base = b << 12;

    // A-fragments for this query tile (row m, contiguous k = 16*half + e, same order as B)
    v16bf a0, a1;
    {
        const bf16_t* ap = h + (long)(base + q0 + m) * C_;
        v8bf lo0 = *(const v8bf*)(ap +      16 * half);
        v8bf hi0 = *(const v8bf*)(ap + 8 + 16 * half);
        v8bf lo1 = *(const v8bf*)(ap + 32 + 16 * half);
        v8bf hi1 = *(const v8bf*)(ap + 40 + 16 * half);
        for (int e = 0; e < 8; ++e) { a0[e] = lo0[e]; a0[8 + e] = hi0[e];
                                      a1[e] = lo1[e]; a1[8 + e] = hi1[e]; }
    }
    float sqm[8];
    for (int v = 0; v < 8; ++v) sqm[v] = sqn[base + q0 + v + 8 * half];

    for (int t = wv; t < 256; t += 16) {          // candidate tiles
        int j0 = t * 16;
        const bf16_t* cp = h + (long)(base + j0 + n) * C_;
        v16bf b0 = *(const v16bf*)(cp +      16 * half);   // B 32x16 layout
        v16bf b1 = *(const v16bf*)(cp + 32 + 16 * half);
        v8f acc = {};
        acc = __builtin_amdgcn_wmma_f32_16x16x32_bf16(false, a0, false, b0, (short)0, acc, false, false);
        acc = __builtin_amdgcn_wmma_f32_16x16x32_bf16(false, a1, false, b1, (short)0, acc, false, false);
        float sj = sqn[base + j0 + n];
        for (int v = 0; v < 8; ++v) {
            int mr = v + 8 * half;                // C/D layout row
            strip[mr * NPTS_ + j0 + n] = sqm[v] + sj - 2.0f * acc[v];
        }
    }
    __syncthreads();

    // selection: wave wv owns query row wv
    topk_select<8>(strip + wv * NPTS_, NPTS_, lane, K_ * dil, dil,
                   idxout + (base + q0 + wv) * K_);
}

// ---------------- block edge-conv: one wave per point, M-dim = 16 neighbors
__global__ __launch_bounds__(256) void k_edgeconv(const bf16_t* __restrict__ hin,
                                                  const int* __restrict__ knn,
                                                  const bf16_t* __restrict__ Wt,   // [64][128]
                                                  const float* __restrict__ bias,  // [64]
                                                  const float* __restrict__ resid, // stride FEATD_
                                                  float* __restrict__ outf,        // stride FEATD_
                                                  bf16_t* __restrict__ hout) {
    int wv = threadIdx.x >> 5, lane = threadIdx.x & 31;
    int p = blockIdx.x * 8 + wv;
    int b = p >> 12;
    int m = lane & 15, half = lane >> 4, n = lane & 15;
    int nb = knn[p * K_ + m];
    const bf16_t* xi = hin + (long)p * C_;
    const bf16_t* xj = hin + (long)((b << 12) + nb) * C_;
    v8f acc[4] = {};
    for (int s = 0; s < 4; ++s) {                 // K = 128 -> 4 k-steps
        v16bf A;
        if (s < 2) {                              // cols 0..63: broadcast x_i
            const bf16_t* ap = xi + 32 * s;
            v8bf lo = *(const v8bf*)(ap +      16 * half);
            v8bf hi = *(const v8bf*)(ap + 8 + 16 * half);
            for (int e = 0; e < 8; ++e) { A[e] = lo[e]; A[8 + e] = hi[e]; }
        } else {                                  // cols 64..127: x_j - x_i
            const bf16_t* ai = xi + 32 * (s - 2);
            const bf16_t* aj = xj + 32 * (s - 2);
            v8bf li = *(const v8bf*)(ai +      16 * half);
            v8bf hi_ = *(const v8bf*)(ai + 8 + 16 * half);
            v8bf lj = *(const v8bf*)(aj +      16 * half);
            v8bf hj = *(const v8bf*)(aj + 8 + 16 * half);
            for (int e = 0; e < 8; ++e) {
                A[e]     = (bf16_t)((float)lj[e] - (float)li[e]);
                A[8 + e] = (bf16_t)((float)hj[e] - (float)hi_[e]);
            }
        }
        for (int t = 0; t < 4; ++t) {             // 4 output tiles of 16
            v16bf B = *(const v16bf*)(Wt + (long)(t * 16 + n) * 128 + 32 * s + 16 * half);
            acc[t] = __builtin_amdgcn_wmma_f32_16x16x32_bf16(false, A, false, B, (short)0, acc[t], false, false);
        }
    }
    for (int t = 0; t < 4; ++t) {
        int c = t * 16 + n;
        float bv = bias[c];
        float mx = -BIGF;
        for (int v = 0; v < 8; ++v) {             // relu then max over neighbors
            float val = acc[t][v] + bv;
            val = val > 0.f ? val : 0.f;
            mx = mx > val ? mx : val;
        }
        float other = __shfl_xor(mx, 16);
        mx = mx > other ? mx : other;
        if (half == 0) {
            float r = mx + resid[(long)p * FEATD_ + c];   // residual
            outf[(long)p * FEATD_ + c] = r;
            hout[(long)p * C_ + c] = (bf16_t)r;
        }
    }
}

// ---------------- head edge-conv: 18-dim feat padded to K=32, one WMMA k-step
__global__ __launch_bounds__(256) void k_edgeconv_head(const float* __restrict__ pos,
                                                       const float* __restrict__ color,
                                                       const int* __restrict__ knn,
                                                       const bf16_t* __restrict__ Wt,  // [64][32]
                                                       const float* __restrict__ bias,
                                                       float* __restrict__ outf,
                                                       bf16_t* __restrict__ hout) {
    int wv = threadIdx.x >> 5, lane = threadIdx.x & 31;
    int p = blockIdx.x * 8 + wv;
    int b = p >> 12;
    int m = lane & 15, half = lane >> 4, n = lane & 15;
    int nb = knn[p * K_ + m];
    int g = (b << 12) + nb;
    float xi[9], xj[9];
    for (int c = 0; c < 3; ++c) { xi[c] = pos[p * 3 + c];     xj[c] = pos[g * 3 + c]; }
    for (int c = 0; c < 6; ++c) { xi[3 + c] = color[p * 6 + c]; xj[3 + c] = color[g * 6 + c]; }
    v16bf A;
    for (int e = 0; e < 16; ++e) {
        int k = 16 * half + e;
        float f = 0.f;
        if (k < 9) f = xi[k]; else if (k < 18) f = xj[k - 9] - xi[k - 9];
        A[e] = (bf16_t)f;
    }
    v8f acc[4] = {};
    for (int t = 0; t < 4; ++t) {
        v16bf B = *(const v16bf*)(Wt + (long)(t * 16 + n) * 32 + 16 * half);
        acc[t] = __builtin_amdgcn_wmma_f32_16x16x32_bf16(false, A, false, B, (short)0, acc[t], false, false);
    }
    for (int t = 0; t < 4; ++t) {
        int c = t * 16 + n;
        float bv = bias[c];
        float mx = -BIGF;
        for (int v = 0; v < 8; ++v) {
            float val = acc[t][v] + bv;
            val = val > 0.f ? val : 0.f;
            mx = mx > val ? mx : val;
        }
        float other = __shfl_xor(mx, 16);
        mx = mx > other ? mx : other;
        if (half == 0) {
            outf[(long)p * FEATD_ + c] = mx;
            hout[(long)p * C_ + c] = (bf16_t)mx;
        }
    }
}

// ---------------- generic bf16 WMMA GEMM: out = [relu](A[M,K] @ Wt[N,K]^T + b)
__global__ __launch_bounds__(256) void k_gemm(const bf16_t* __restrict__ A,
                                              const bf16_t* __restrict__ Wt,
                                              const float* __restrict__ bias,
                                              float* __restrict__ outf,
                                              bf16_t* __restrict__ outb,
                                              int Kdim, int Npad, int Nwrite, int ldF, int relu) {
    int wv = threadIdx.x >> 5, lane = threadIdx.x & 31;
    int m = lane & 15, half = lane >> 4, n = lane & 15;
    int mt = blockIdx.x;
    int nt = blockIdx.y * 8 + wv;
    if (nt * 16 >= Npad) return;
    long arow = (long)(mt * 16 + m) * Kdim;
    long brow = (long)(nt * 16 + n) * Kdim;
    v8f acc = {};
    int ks = Kdim >> 5;
    for (int s = 0; s < ks; ++s) {
        if (s + 4 < ks) __builtin_prefetch(Wt + brow + 32 * (s + 4), 0, 0);
        const bf16_t* ap = A + arow + 32 * s;
        v8bf lo = *(const v8bf*)(ap +      16 * half);
        v8bf hi = *(const v8bf*)(ap + 8 + 16 * half);
        v16bf Av;
        for (int e = 0; e < 8; ++e) { Av[e] = lo[e]; Av[8 + e] = hi[e]; }
        v16bf Bv = *(const v16bf*)(Wt + brow + 32 * s + 16 * half);
        acc = __builtin_amdgcn_wmma_f32_16x16x32_bf16(false, Av, false, Bv, (short)0, acc, false, false);
    }
    int c = nt * 16 + n;
    float bv = bias ? bias[c] : 0.f;
    for (int v = 0; v < 8; ++v) {
        long r = (long)(mt * 16 + v + 8 * half);
        float val = acc[v] + bv;
        if (relu) val = val > 0.f ? val : 0.f;
        if (outb) outb[r * Npad + c] = (bf16_t)val;
        if (outf && c < Nwrite) outf[r * ldF + c] = val;
    }
}

// ---------------------------------------------------------------------------
extern "C" void kernel_launch(void* const* d_in, const int* in_sizes, int n_in,
                              void* d_out, int out_size, void* d_ws, size_t ws_size,
                              hipStream_t stream) {
    const float* pos      = (const float*)d_in[0];
    const float* color    = (const float*)d_in[1];
    // d_in[2] = batch (derived from index, unused)
    const float* w_head   = (const float*)d_in[3];
    const float* b_head   = (const float*)d_in[4];
    const float* w_blocks = (const float*)d_in[5];   // (7,128,64)
    const float* b_blocks = (const float*)d_in[6];   // (7,64)
    const float* w1       = (const float*)d_in[7];   // (512,512)
    const float* b1       = (const float*)d_in[8];
    const float* w2       = (const float*)d_in[9];   // (512,256)
    const float* b2       = (const float*)d_in[10];
    const float* w3       = (const float*)d_in[11];  // (256,13)
    const float* b3       = (const float*)d_in[12];
    float* out            = (float*)d_out;           // (16384,13)

    // ---- workspace carve-up (~82 MB) ----
    char* ws = (char*)d_ws;
    size_t off = 0;
    auto alloc = [&](size_t bytes) { char* p = ws + off; off += (bytes + 255) & ~(size_t)255; return p; };
    float*  feats  = (float*) alloc((size_t)NTOT_ * FEATD_ * 4);
    bf16_t* featsb = (bf16_t*)alloc((size_t)NTOT_ * FEATD_ * 2);
    bf16_t* hbfA   = (bf16_t*)alloc((size_t)NTOT_ * C_ * 2);
    bf16_t* hbfB   = (bf16_t*)alloc((size_t)NTOT_ * C_ * 2);
    float*  sqn    = (float*) alloc((size_t)NTOT_ * 4);
    int*    knnix  = (int*)   alloc((size_t)NTOT_ * K_ * 4);
    bf16_t* WtH    = (bf16_t*)alloc(64 * 32 * 2);
    bf16_t* WtB    = (bf16_t*)alloc((size_t)NB_ * 64 * 128 * 2);
    bf16_t* Wt1    = (bf16_t*)alloc(512 * 512 * 2);
    bf16_t* Wt2    = (bf16_t*)alloc(256 * 512 * 2);
    bf16_t* Wt3    = (bf16_t*)alloc(16 * 256 * 2);
    bf16_t* f1b    = (bf16_t*)alloc((size_t)NTOT_ * 512 * 2);
    bf16_t* f2b    = (bf16_t*)alloc((size_t)NTOT_ * 256 * 2);
    (void)ws_size; (void)n_in; (void)in_sizes; (void)out_size;

    // ---- weight prep ----
    k_prep_wt<<<8, 256, 0, stream>>>(w_head, WtH, 18, 32, 64, 64);
    for (int i = 0; i < NB_; ++i)
        k_prep_wt<<<32, 256, 0, stream>>>(w_blocks + (size_t)i * 128 * 64, WtB + (size_t)i * 64 * 128,
                                          128, 128, 64, 64);
    k_prep_wt<<<1024, 256, 0, stream>>>(w1, Wt1, 512, 512, 512, 512);
    k_prep_wt<<<512, 256, 0, stream>>>(w2, Wt2, 512, 512, 256, 256);
    k_prep_wt<<<16, 256, 0, stream>>>(w3, Wt3, 256, 256, 13, 16);

    // ---- head: KNN on positions + edge-conv (18->64) ----
    k_knn_head<<<NTOT_, 32, 0, stream>>>(pos, knnix);
    k_edgeconv_head<<<NTOT_ / 8, 256, 0, stream>>>(pos, color, knnix, WtH, b_head, feats, hbfA);

    // ---- 7 residual GCN blocks ----
    bf16_t* cur = hbfA; bf16_t* nxt = hbfB;
    for (int i = 0; i < NB_; ++i) {
        k_sqnorm<<<NTOT_ / 256, 256, 0, stream>>>(cur, sqn);
        k_knn_feat<<<1024, 512, 16 * NPTS_ * sizeof(float), stream>>>(cur, sqn, knnix, i + 1);
        k_edgeconv<<<NTOT_ / 8, 256, 0, stream>>>(cur, knnix, WtB + (size_t)i * 64 * 128,
                                                  b_blocks + (size_t)i * 64,
                                                  feats + (size_t)i * C_,        // residual = stage i
                                                  feats + (size_t)(i + 1) * C_,  // write stage i+1
                                                  nxt);
        bf16_t* t = cur; cur = nxt; nxt = t;
    }

    // ---- classifier MLP ----
    k_f2bf<<<(NTOT_ * FEATD_) / 256, 256, 0, stream>>>(feats, featsb, NTOT_ * FEATD_);
    dim3 blk(256);
    k_gemm<<<dim3(NTOT_ / 16, 4), blk, 0, stream>>>(featsb, Wt1, b1, nullptr, f1b, 512, 512, 512, 512, 1);
    k_gemm<<<dim3(NTOT_ / 16, 2), blk, 0, stream>>>(f1b,    Wt2, b2, nullptr, f2b, 512, 256, 256, 256, 1);
    k_gemm<<<dim3(NTOT_ / 16, 1), blk, 0, stream>>>(f2b,    Wt3, b3, out, nullptr, 256, 16, 13, 13, 0);
}
